// MultiHeadAttention_83854941487661
// MI455X (gfx1250) — matrix operand order repaired and checked by reading, not compiled, not run on hardware
//
#include <hip/hip_runtime.h>

// Problem constants (fixed by reference setup_inputs)
#define BB 2
#define TT 2048
#define DD 1024
#define HH 16
#define HD 64
#define MM (BB*TT)      // 4096 rows for the projections

typedef __attribute__((ext_vector_type(16))) _Float16 v16h;
typedef __attribute__((ext_vector_type(8)))  _Float16 v8h;
typedef __attribute__((ext_vector_type(8)))  float    v8f;
typedef int i32x4 __attribute__((vector_size(16)));   // matches builtin param types

#define WMMA16(a,b,c) __builtin_amdgcn_wmma_f32_16x16x32_f16(false,(a),false,(b),(short)0,(c),false,false)

// ---- CDNA5 async global->LDS copy (ASYNCcnt path), guarded with fallback ----
#if __has_builtin(__builtin_amdgcn_global_load_async_to_lds_b128)
#define HAVE_ASYNC_LDS 1
#endif

__device__ __forceinline__ void async_copy16(const void* gp, void* lp) {
#if defined(HAVE_ASYNC_LDS)
    __builtin_amdgcn_global_load_async_to_lds_b128(
        (__attribute__((address_space(1))) i32x4*)(void*)gp,
        (__attribute__((address_space(3))) i32x4*)lp, 0, 0);
#else
    *(v8h*)lp = *(const v8h*)gp;
#endif
}

__device__ __forceinline__ void wait_async() {
#if defined(HAVE_ASYNC_LDS)
#if __has_builtin(__builtin_amdgcn_s_wait_asynccnt)
    __builtin_amdgcn_s_wait_asynccnt(0);
#else
    asm volatile("s_wait_asynccnt 0x0" ::: "memory");
#endif
#endif
}

// ---- CDNA5 Tensor Data Mover (TENSORcnt path), guarded ----
#if __has_builtin(__builtin_amdgcn_tensor_load_to_lds) && __has_builtin(__builtin_amdgcn_s_wait_tensorcnt)
#define HAVE_TDM 1
typedef unsigned int u32x4 __attribute__((vector_size(16)));
typedef int          i32x8 __attribute__((vector_size(32)));

// 2D tile load: tile_h rows of tile_w f16 elements, global row stride = row_stride
// elements, into contiguous LDS at ldst. Descriptor per CDNA5 ISA.
__device__ __forceinline__ void tdm_load_tile_f16(const void* gsrc, void* ldst,
                                                  unsigned tile_w, unsigned tile_h,
                                                  unsigned row_stride) {
    unsigned lds_off = (unsigned)(unsigned long long)
        (__attribute__((address_space(3))) void*)ldst;
    unsigned long long ga = (unsigned long long)gsrc;   // byte address
    u32x4 g0 = {0u, 0u, 0u, 0u};
    g0[0] = 1u;                                          // count=1, user mode
    g0[1] = lds_off;                                     // lds_addr (bytes)
    g0[2] = (unsigned)(ga & 0xffffffffu);                // global_addr[31:0]
    g0[3] = (unsigned)((ga >> 32) & 0x01ffffffu)         // global_addr[56:32]
          | (2u << 30);                                  // type = 2 ("image")
    i32x8 g1 = {0,0,0,0,0,0,0,0};
    g1[0] = (int)(1u << 16);                             // data_size=1 (2 bytes)
    g1[1] = (int)((row_stride & 0xffffu) << 16);         // tensor_dim0[15:0]
    g1[2] = (int)(((row_stride >> 16) & 0xffffu)         // tensor_dim0[31:16]
          | ((tile_h & 0xffffu) << 16));                 // tensor_dim1[15:0]
    g1[3] = (int)(((tile_h >> 16) & 0xffffu)             // tensor_dim1[31:16]
          | ((tile_w & 0xffffu) << 16));                 // tile_dim0
    g1[4] = (int)(tile_h & 0xffffu);                     // tile_dim1 (tile_dim2=0)
    g1[5] = (int)row_stride;                             // tensor_dim0_stride[31:0]
    g1[6] = 0;                                           // stride[47:32]=0, dim1_stride=0
    g1[7] = 0;
    i32x4 gz4 = {0, 0, 0, 0};                            // groups 2/3 unused (2D)
    i32x8 gz8 = {0, 0, 0, 0, 0, 0, 0, 0};                // 6-arg form extra group
    __builtin_amdgcn_tensor_load_to_lds(g0, g1, gz4, gz4, gz8, 0);
}

__device__ __forceinline__ void wait_tensor() {
    __builtin_amdgcn_s_wait_tensorcnt(0);
}
#endif

__device__ __forceinline__ v16h load_frag_pair(const _Float16* p0, const _Float16* p1) {
    v8h lo = *(const v8h*)p0;
    v8h hi = *(const v8h*)p1;
    v16h r;
#pragma unroll
    for (int i = 0; i < 8; ++i) { r[i] = lo[i]; r[8 + i] = hi[i]; }
    return r;
}

// ---------------------------------------------------------------------------
// Generic 64x64-tile GEMM, C = A[M,K] * B[K,N]; 128 threads = 4 waves,
// each wave computes a 32x32 subtile as 2x2 WMMA 16x16 tiles, K staged 32 wide.
//   A_F16   : A is f16 (ctx in ws, TDM-staged) vs f32 (x input, converted)
//   MODE_OUT: write f32 row-major + bias (out proj) vs f16 head-major (QKV)
// ---------------------------------------------------------------------------
template <bool A_F16, bool MODE_OUT>
__global__ __launch_bounds__(128) void gemm64_kernel(
    const void* __restrict__ Aptr,
    const float* __restrict__ B0, const float* __restrict__ B1, const float* __restrict__ B2,
    void* __restrict__ O0, void* __restrict__ O1, void* __restrict__ O2,
    const float* __restrict__ bias)
{
    constexpr int M = MM, N = DD, K = DD;
    const int nb = blockIdx.x;   // N/64
    const int mb = blockIdx.y;   // M/64
    const int mat = blockIdx.z;  // weight / destination select
    const float* Bg = (mat == 0) ? B0 : ((mat == 1) ? B1 : B2);
    void* Out = (mat == 0) ? O0 : ((mat == 1) ? O1 : O2);

    __shared__ alignas(16) _Float16 As[64][32];
    __shared__ alignas(16) _Float16 Bt[64][32];   // transposed: Bt[n][k]

    const int tid  = threadIdx.x;
    const int wave = tid >> 5;
    const int lane = tid & 31;
    const int half = lane >> 4;
    const int l16  = lane & 15;
    const int wr   = (wave >> 1) * 32;  // wave row offset in 64x64 tile
    const int wc   = (wave & 1) * 32;   // wave col offset
    const int mBase = mb * 64, nBase = nb * 64;

    const v8f zero8 = {0.f,0.f,0.f,0.f,0.f,0.f,0.f,0.f};
    v8f acc[2][2];
#pragma unroll
    for (int i = 0; i < 2; ++i)
#pragma unroll
        for (int j = 0; j < 2; ++j) acc[i][j] = zero8;

    for (int k0 = 0; k0 < K; k0 += 32) {
        __syncthreads();
        // ---- stage A tile (64 x 32) into LDS as f16 ----
        if constexpr (A_F16) {
            const _Float16* Ag = (const _Float16*)Aptr;
#if defined(HAVE_TDM)
            // one TDM descriptor moves the whole 64x32 f16 tile (strided rows)
            if (wave == 0)
                tdm_load_tile_f16(&Ag[(size_t)mBase * K + k0], (void*)&As[0][0],
                                  /*tile_w=*/32, /*tile_h=*/64, /*row_stride=*/K);
#else
#pragma unroll
            for (int j = 0; j < 2; ++j) {
                int f = tid + j * 128;      // 256 chunks of 8 halves
                int r = f >> 2, c = f & 3;
                async_copy16(&Ag[(size_t)(mBase + r) * K + k0 + c * 8], &As[r][c * 8]);
            }
#endif
        } else {
            const float* Ag = (const float*)Aptr;
#pragma unroll
            for (int j = 0; j < 4; ++j) {
                int f = tid + j * 128;      // 512 float4
                int r = f >> 3, c = f & 7;
                const float4 v = *(const float4*)&Ag[(size_t)(mBase + r) * K + k0 + c * 4];
                As[r][c * 4 + 0] = (_Float16)v.x;
                As[r][c * 4 + 1] = (_Float16)v.y;
                As[r][c * 4 + 2] = (_Float16)v.z;
                As[r][c * 4 + 3] = (_Float16)v.w;
            }
        }
        // ---- stage B tile (32 x 64) transposed into LDS as f16 ----
#pragma unroll
        for (int j = 0; j < 4; ++j) {
            int f = tid + j * 128;          // 512 float4: 32 k-rows x 16 float4
            int kr = f >> 4, c = f & 15;
            const float4 v = *(const float4*)&Bg[(size_t)(k0 + kr) * N + nBase + c * 4];
            Bt[c * 4 + 0][kr] = (_Float16)v.x;
            Bt[c * 4 + 1][kr] = (_Float16)v.y;
            Bt[c * 4 + 2][kr] = (_Float16)v.z;
            Bt[c * 4 + 3][kr] = (_Float16)v.w;
        }
        if constexpr (A_F16) {
#if defined(HAVE_TDM)
            if (wave == 0) wait_tensor();
#else
            wait_async();
#endif
        }
        __syncthreads();

        // ---- fragments + WMMA ----
        v16h aF[2], bF[2];
#pragma unroll
        for (int i = 0; i < 2; ++i) {
            int r = wr + i * 16 + l16;
            aF[i] = load_frag_pair(&As[r][half * 8], &As[r][16 + half * 8]);
        }
#pragma unroll
        for (int j = 0; j < 2; ++j) {
            int n = wc + j * 16 + l16;
            bF[j] = load_frag_pair(&Bt[n][half * 8], &Bt[n][16 + half * 8]);
        }
#pragma unroll
        for (int i = 0; i < 2; ++i)
#pragma unroll
            for (int j = 0; j < 2; ++j)
                acc[i][j] = WMMA16(aF[i], bF[j], acc[i][j]);
    }

    // ---- store: C layout lane l16 = N col, half selects M+8, vgpr v = M row ----
#pragma unroll
    for (int i = 0; i < 2; ++i)
#pragma unroll
        for (int j = 0; j < 2; ++j)
#pragma unroll
            for (int v = 0; v < 8; ++v) {
                int rg = mBase + wr + i * 16 + half * 8 + v;
                int cg = nBase + wc + j * 16 + l16;
                float val = acc[i][j][v];
                if constexpr (MODE_OUT) {
                    ((float*)Out)[(size_t)rg * N + cg] = val + bias[cg];
                } else {
                    // head-major: [b][h][t][hd]
                    int b = rg >> 11, t = rg & (TT - 1);
                    int h = cg >> 6,  d = cg & (HD - 1);
                    ((_Float16*)Out)[(((size_t)b * HH + h) * TT + t) * HD + d] = (_Float16)val;
                }
            }
}

// ---------------------------------------------------------------------------
// Flash attention: one block per (b, h, 64-row query tile), 4 waves x 16 rows.
// Mask with -1e9 BEFORE the 1/sqrt(hd)=0.125 scale (matches reference).
// K tiles staged via async global->LDS (ASYNCcnt); V transposed via VALU.
// ---------------------------------------------------------------------------
__global__ __launch_bounds__(128) void attn_kernel(
    const _Float16* __restrict__ Q, const _Float16* __restrict__ Kg,
    const _Float16* __restrict__ Vg, _Float16* __restrict__ ctx)
{
    const int qt = blockIdx.x;   // T/64 tiles
    const int h  = blockIdx.y;
    const int b  = blockIdx.z;
    const size_t headOff = ((size_t)b * HH + h) * TT * HD;
    const _Float16* Qh = Q  + headOff;
    const _Float16* Kh = Kg + headOff;
    const _Float16* Vh = Vg + headOff;

    __shared__ alignas(16) _Float16 Ks[32][64];      // [key][d]
    __shared__ alignas(16) _Float16 Vt[64][32];      // [d][key] (transposed)
    __shared__ alignas(16) _Float16 Ps[4][16][32];   // per-wave P scratch

    const int tid  = threadIdx.x;
    const int wave = tid >> 5;
    const int lane = tid & 31;
    const int half = lane >> 4;
    const int l16  = lane & 15;
    const int qBase = qt * 64 + wave * 16;

    // Q fragments (K-dim = hd = 64 -> 2 fragments), straight from global
    v16h qF[2];
    {
        const _Float16* qrow = Qh + (size_t)(qBase + l16) * HD;
        qF[0] = load_frag_pair(qrow + half * 8,      qrow + 16 + half * 8);
        qF[1] = load_frag_pair(qrow + 32 + half * 8, qrow + 48 + half * 8);
    }

    const v8f zero8 = {0.f,0.f,0.f,0.f,0.f,0.f,0.f,0.f};
    float mrow[8], lrow[8];
    v8f O[4];
#pragma unroll
    for (int v = 0; v < 8; ++v) { mrow[v] = -3.0e38f; lrow[v] = 0.f; }
#pragma unroll
    for (int j = 0; j < 4; ++j) O[j] = zero8;

    const int numKT = qt * 2 + 2;   // causal: keys < qt*64+64
    for (int kt = 0; kt < numKT; ++kt) {
        const int kBase = kt * 32;
        __syncthreads();
        // stage K tile via async global->LDS; V tile transposed via registers
#pragma unroll
        for (int j = 0; j < 2; ++j) {
            int f = tid + j * 128;          // 256 chunks of 8 halves
            int r = f >> 3, c = f & 7;
            async_copy16(&Kh[(size_t)(kBase + r) * HD + c * 8], &Ks[r][c * 8]);
            v8h vv = *(const v8h*)&Vh[(size_t)(kBase + r) * HD + c * 8];
#pragma unroll
            for (int e = 0; e < 8; ++e) Vt[c * 8 + e][r] = vv[e];
        }
        wait_async();
        __syncthreads();

        // S = Q * K^T  (two 16x16 key subtiles)
        v8f sAcc[2];
#pragma unroll
        for (int sub = 0; sub < 2; ++sub) {
            sAcc[sub] = zero8;
#pragma unroll
            for (int ks = 0; ks < 2; ++ks) {
                int key = sub * 16 + l16;
                v16h kF = load_frag_pair(&Ks[key][ks * 32 + half * 8],
                                         &Ks[key][ks * 32 + 16 + half * 8]);
                sAcc[sub] = WMMA16(qF[ks], kF, sAcc[sub]);
            }
        }

        // mask + scale + tile row-max (16-lane reduction within half)
        float mt[8];
#pragma unroll
        for (int v = 0; v < 8; ++v) {
            int qg = qBase + half * 8 + v;
#pragma unroll
            for (int sub = 0; sub < 2; ++sub) {
                int kg = kBase + sub * 16 + l16;
                float s = sAcc[sub][v];
                if (kg > qg) s = -1.0e9f;   // mask before scale (reference)
                s *= 0.125f;
                sAcc[sub][v] = s;
            }
            float m = fmaxf(sAcc[0][v], sAcc[1][v]);
            m = fmaxf(m, __shfl_xor(m, 1));
            m = fmaxf(m, __shfl_xor(m, 2));
            m = fmaxf(m, __shfl_xor(m, 4));
            m = fmaxf(m, __shfl_xor(m, 8));
            mt[v] = m;
        }

        // online softmax update
#pragma unroll
        for (int v = 0; v < 8; ++v) {
            float mnew  = fmaxf(mrow[v], mt[v]);
            float alpha = __expf(mrow[v] - mnew);
            float psum  = 0.f;
#pragma unroll
            for (int sub = 0; sub < 2; ++sub) {
                float p = __expf(sAcc[sub][v] - mnew);
                sAcc[sub][v] = p;
                psum += p;
            }
            psum += __shfl_xor(psum, 1);
            psum += __shfl_xor(psum, 2);
            psum += __shfl_xor(psum, 4);
            psum += __shfl_xor(psum, 8);
            lrow[v] = lrow[v] * alpha + psum;
            mrow[v] = mnew;
#pragma unroll
            for (int j = 0; j < 4; ++j) O[j][v] *= alpha;
        }

        // C-layout -> A-layout via per-wave LDS scratch
#pragma unroll
        for (int v = 0; v < 8; ++v)
#pragma unroll
            for (int sub = 0; sub < 2; ++sub)
                Ps[wave][half * 8 + v][sub * 16 + l16] = (_Float16)sAcc[sub][v];
        __syncthreads();

        v16h pF = load_frag_pair(&Ps[wave][l16][half * 8], &Ps[wave][l16][16 + half * 8]);
#pragma unroll
        for (int j = 0; j < 4; ++j) {
            v16h vF = load_frag_pair(&Vt[j * 16 + l16][half * 8],
                                     &Vt[j * 16 + l16][16 + half * 8]);
            O[j] = WMMA16(pF, vF, O[j]);
        }
    }

    // epilogue: O / l  -> ctx [b][t][h*64+d] f16
#pragma unroll
    for (int v = 0; v < 8; ++v) {
        float inv = 1.0f / lrow[v];
        int tg = qBase + half * 8 + v;
#pragma unroll
        for (int j = 0; j < 4; ++j)
            ctx[((size_t)b * TT + tg) * DD + h * HD + j * 16 + l16] =
                (_Float16)(O[j][v] * inv);
    }
}

// ---------------------------------------------------------------------------
extern "C" void kernel_launch(void* const* d_in, const int* in_sizes, int n_in,
                              void* d_out, int out_size, void* d_ws, size_t ws_size,
                              hipStream_t stream)
{
    const float* x  = (const float*)d_in[0];
    const float* Wq = (const float*)d_in[1];
    const float* Wk = (const float*)d_in[2];
    const float* Wv = (const float*)d_in[3];
    const float* Wo = (const float*)d_in[4];
    const float* bo = (const float*)d_in[5];

    const size_t headElems = (size_t)BB * HH * TT * HD;  // 4M
    _Float16* Qw = (_Float16*)d_ws;
    _Float16* Kw = Qw + headElems;
    _Float16* Vw = Kw + headElems;
    _Float16* Cw = Vw + headElems;                       // ctx [b*t][1024]

    // 1) QKV projections (z selects weight/destination)
    gemm64_kernel<false, false><<<dim3(DD / 64, MM / 64, 3), 128, 0, stream>>>(
        (const void*)x, Wq, Wk, Wv, (void*)Qw, (void*)Kw, (void*)Vw, nullptr);

    // 2) causal flash attention
    attn_kernel<<<dim3(TT / 64, HH, BB), 128, 0, stream>>>(Qw, Kw, Vw, Cw);

    // 3) output projection + bias -> f32
    gemm64_kernel<true, true><<<dim3(DD / 64, MM / 64, 1), 128, 0, stream>>>(
        (const void*)Cw, Wo, Wo, Wo, d_out, d_out, d_out, bo);

    (void)in_sizes; (void)n_in; (void)out_size; (void)ws_size;
}
